// DSASNetVoxelBackBone8x_77713138253943
// MI455X (gfx1250) — matrix operand order repaired and checked by reading, not compiled, not run on hardware
//
#include <hip/hip_runtime.h>
#include <hip/hip_bf16.h>

typedef __attribute__((ext_vector_type(16))) _Float16 v16h;
typedef __attribute__((ext_vector_type(8)))  _Float16 v8h;
typedef __attribute__((ext_vector_type(8)))  float    v8f;

// ---------------------------------------------------------------------------
// Utility kernels
// ---------------------------------------------------------------------------
__global__ void zero_f32(float* __restrict__ p, long long nf) {
  long long i = (long long)blockIdx.x * blockDim.x + threadIdx.x;
  if (i < nf) p[i] = 0.0f;
}

// voxel features [n0,4] f32 -> padded half [(n0+1), 32], zero pad + sentinel row
__global__ void feat0_prep(const float* __restrict__ vox, _Float16* __restrict__ f0,
                           int n0) {
  long long i = (long long)blockIdx.x * blockDim.x + threadIdx.x;
  long long tot = (long long)(n0 + 1) * 32;
  if (i >= tot) return;
  long long row = i >> 5;
  int c = (int)(i & 31);
  float v = (row < n0 && c < 4) ? vox[row * 4 + c] : 0.0f;
  f0[i] = (_Float16)v;
}

// weights [K,Cin,Cout] f32 -> transposed padded half [K,Cout,Cinp]
__global__ void wprep(const float* __restrict__ w, _Float16* __restrict__ wt,
                      int K, int Cin, int Cout, int Cinp) {
  long long i = (long long)blockIdx.x * blockDim.x + threadIdx.x;
  long long tot = (long long)K * Cout * Cinp;
  if (i >= tot) return;
  int ci = (int)(i % Cinp);
  long long t2 = i / Cinp;
  int co = (int)(t2 % Cout);
  int k  = (int)(t2 / Cout);
  float v = (ci < Cin) ? w[((size_t)k * Cin + ci) * Cout + co] : 0.0f;
  wt[i] = (_Float16)v;
}

// ---------------------------------------------------------------------------
// Sparse conv: gather -> WMMA f16 GEMM (fp32 accum) -> scatter atomic add
// feats: [(bound+1), CINP] f16, rows >= n_in zeroed (incl. sentinel row n_in)
// wt:    [K, COUT, CINP] f16 (pre-transposed, K-padded)
// A layout per ISA: lane l row=l&15; elems 0..7 = K[seg..seg+7],
//                   elems 8..15 = K[16+seg..16+seg+7], seg=(l>=16?8:0)
// B layout: lane l col=l&15; same K mapping as A, bseg=(l<16?0:8)
// C/D: lane l col=l&15; VGPR r -> row M = r + (l>=16?8:0)
// ---------------------------------------------------------------------------
template <int CIN, int COUT>
__global__ __launch_bounds__(256) void spconv_gemm_t(
    const _Float16* __restrict__ feats,
    const _Float16* __restrict__ wt,
    const int* __restrict__ rin, const int* __restrict__ rout,
    float* __restrict__ acc,
    const int* __restrict__ nin_dev,
    int K, int L, int tilesPerK)
{
  constexpr int CINP   = (CIN + 31) & ~31;
  constexpr int KSTEPS = CINP / 32;

  int wid  = blockIdx.x * (blockDim.x >> 5) + (threadIdx.x >> 5);
  int lane = threadIdx.x & 31;
  if (wid >= K * tilesPerK) return;       // wave-uniform: EXEC stays full

  int k = wid / tilesPerK;
  int t = wid - k * tilesPerK;
  int base = t * 16;
  int n_in = *nin_dev;

  int colLo = lane & 15;
  int mBase = (lane >= 16) ? 8 : 0;
  int seg   = (lane >= 16) ? 8 : 0;
  int bseg  = (lane < 16) ? 0 : 8;

  // ---- gather index (branchless clamp + sentinel select) ----
  int pos  = base + colLo;
  int posC = (pos < L) ? pos : (L - 1);
  int g    = rin[(size_t)k * L + posC];
  if (pos >= L) g = n_in;                 // sentinel -> zero row
  const _Float16* frow = feats + (size_t)g * CINP + seg;

  // ---- scatter indices: hoisted, one clause of 8 loads ----
  const int* rbase = rout + (size_t)k * L;
  int op[8];
#pragma unroll
  for (int r = 0; r < 8; ++r) {
    int p2  = base + mBase + r;
    int p2c = (p2 < L) ? p2 : (L - 1);
    op[r] = rbase[p2c];
  }
  float* aptr[8];
#pragma unroll
  for (int r = 0; r < 8; ++r)
    aptr[r] = acc + (size_t)((unsigned)op[r]) * COUT + colLo;

  // ---- A fragments: compile-time offsets off one base pointer ----
  v16h afrag[KSTEPS];
#pragma unroll
  for (int ks = 0; ks < KSTEPS; ++ks) {
    v8h lo = *(const v8h*)(frow + ks * 32);
    v8h hi = *(const v8h*)(frow + ks * 32 + 16);
    v16h a;
#pragma unroll
    for (int e = 0; e < 8; ++e) { a[e] = lo[e]; a[8 + e] = hi[e]; }
    afrag[ks] = a;
  }

  // ---- B base: all (nt, ks) offsets become instruction immediates ----
  const _Float16* wbase = wt + ((size_t)k * COUT + colLo) * CINP + bseg;

#pragma unroll
  for (int nt = 0; nt < COUT / 16; ++nt) {
    v8f c = {};
#pragma unroll
    for (int ks = 0; ks < KSTEPS; ++ks) {
      v8h blo = *(const v8h*)(wbase + nt * 16 * CINP + ks * 32);
      v8h bhi = *(const v8h*)(wbase + nt * 16 * CINP + ks * 32 + 16);
      v16h b;
#pragma unroll
      for (int e = 0; e < 8; ++e) { b[e] = blo[e]; b[8 + e] = bhi[e]; }
      c = __builtin_amdgcn_wmma_f32_16x16x32_f16(false, afrag[ks], false, b,
                                                 (short)0, c, false, false);
    }
    // Unconditional scatter: clamped rows have zero A-row -> c[r] == 0.
    // nt*16 folds into the atomic's immediate offset.
#pragma unroll
    for (int r = 0; r < 8; ++r) {
      atomicAdd(aptr[r] + nt * 16, c[r]);
    }
  }
}

// ---------------------------------------------------------------------------
// BN statistics: per-channel sum / sumsq, LDS reduce then atomics.
// COUT power of two, <= 128.
// ---------------------------------------------------------------------------
__global__ __launch_bounds__(256) void bn_stats(
    const float* __restrict__ x, float* __restrict__ stats,
    const int* __restrict__ n_dev, int Cout)
{
  __shared__ float ssum[256];
  __shared__ float ssq[256];
  int n   = *n_dev;
  int tid = threadIdx.x;
  int ch  = tid & (Cout - 1);
  int g   = tid / Cout;
  int G   = 256 / Cout;

  float s = 0.0f, q = 0.0f;
  for (long long row = (long long)blockIdx.x * G + g; row < n;
       row += (long long)gridDim.x * G) {
    float v = x[row * Cout + ch];
    s += v; q += v * v;
  }
  ssum[tid] = s; ssq[tid] = q;
  __syncthreads();
  for (int off = 128; off >= Cout; off >>= 1) {
    if (tid < off) { ssum[tid] += ssum[tid + off]; ssq[tid] += ssq[tid + off]; }
    __syncthreads();
  }
  if (tid < Cout) {
    atomicAdd(&stats[tid], ssum[tid]);
    atomicAdd(&stats[Cout + tid], ssq[tid]);
  }
}

// ---------------------------------------------------------------------------
// BN apply + ReLU -> padded f16 feats (rows >= n and pad cols -> 0)
// ---------------------------------------------------------------------------
template <int COUT>
__global__ void bn_apply_half(
    const float* __restrict__ x, const float* __restrict__ stats,
    const float* __restrict__ gamma, const float* __restrict__ beta,
    _Float16* __restrict__ y, const int* __restrict__ n_dev,
    long long rowsTotal)
{
  constexpr int CP = (COUT + 31) & ~31;
  long long i = (long long)blockIdx.x * blockDim.x + threadIdx.x;
  long long tot = rowsTotal * CP;
  if (i >= tot) return;
  long long row = i / CP;
  int ch = (int)(i - row * CP);
  int n = *n_dev;
  float out = 0.0f;
  if (row < (long long)n && ch < COUT) {
    float inv_n = 1.0f / (float)n;
    float mu  = stats[ch] * inv_n;
    float var = stats[COUT + ch] * inv_n - mu * mu;
    float v = (x[row * COUT + ch] - mu) * rsqrtf(var + 1e-3f) * gamma[ch] + beta[ch];
    out = v > 0.0f ? v : 0.0f;
  }
  y[i] = (_Float16)out;
}

// BN apply + ReLU -> dense f32 (final outputs into d_out)
__global__ void bn_apply_f32(
    const float* __restrict__ x, const float* __restrict__ stats,
    const float* __restrict__ gamma, const float* __restrict__ beta,
    float* __restrict__ y, const int* __restrict__ n_dev,
    int Cout, long long rowsTotal)
{
  long long i = (long long)blockIdx.x * blockDim.x + threadIdx.x;
  long long tot = rowsTotal * Cout;
  if (i >= tot) return;
  int n = *n_dev;
  long long row = i / Cout;
  int ch = (int)(i - row * Cout);
  float out = 0.0f;
  if (row < (long long)n) {
    float inv_n = 1.0f / (float)n;
    float mu  = stats[ch] * inv_n;
    float var = stats[Cout + ch] * inv_n - mu * mu;
    float v = (x[i] - mu) * rsqrtf(var + 1e-3f) * gamma[ch] + beta[ch];
    out = v > 0.0f ? v : 0.0f;
  }
  y[i] = out;
}

// ---------------------------------------------------------------------------
// point_coords
// ---------------------------------------------------------------------------
__global__ void point_coords_k(const int* __restrict__ c1,
                               const float* __restrict__ vs,
                               const float* __restrict__ pr,
                               float* __restrict__ out, int n1)
{
  int i = blockIdx.x * blockDim.x + threadIdx.x;
  if (i >= n1) return;
  out[(size_t)i * 4 + 0] = (float)c1[(size_t)i * 4 + 0];
#pragma unroll
  for (int j = 0; j < 3; ++j) {
    out[(size_t)i * 4 + 1 + j] =
        ((float)c1[(size_t)i * 4 + (3 - j)] + 0.5f) * (vs[j] * 2.0f) + pr[j];
  }
}

// ---------------------------------------------------------------------------
// Host orchestration
// ---------------------------------------------------------------------------
struct RuleH { const int* rin; const int* rout; const int* nout; int K; int L; };

template <int CIN, int COUT>
static void runLayer(const _Float16* fin, const int* nin_dev, const RuleH& r,
                     const float* w, const float* gamma, const float* beta,
                     float* acc, float* stats, _Float16* wt,
                     int boundOut, _Float16* destH, float* destF,
                     long long destRows, hipStream_t stream)
{
  constexpr int CINP = (CIN + 31) & ~31;
  long long wtN = (long long)r.K * COUT * CINP;
  wprep<<<dim3((unsigned)((wtN + 255) / 256)), 256, 0, stream>>>(
      w, wt, r.K, CIN, COUT, CINP);

  long long accN = (long long)(boundOut + 1) * COUT;
  zero_f32<<<dim3((unsigned)((accN + 255) / 256)), 256, 0, stream>>>(acc, accN);
  zero_f32<<<dim3(1), 256, 0, stream>>>(stats, 2 * COUT);

  int tiles = (r.L + 15) / 16;
  long long blocks = ((long long)r.K * tiles * 32 + 255) / 256;
  spconv_gemm_t<CIN, COUT><<<dim3((unsigned)blocks), 256, 0, stream>>>(
      fin, wt, r.rin, r.rout, acc, nin_dev, r.K, r.L, tiles);

  int G = 256 / COUT;
  long long sb = ((long long)boundOut + G - 1) / G;
  if (sb < 1) sb = 1;
  if (sb > 4096) sb = 4096;
  bn_stats<<<dim3((unsigned)sb), 256, 0, stream>>>(acc, stats, r.nout, COUT);

  if (destH) {
    constexpr int CP = (COUT + 31) & ~31;
    long long tot = destRows * CP;
    bn_apply_half<COUT><<<dim3((unsigned)((tot + 255) / 256)), 256, 0, stream>>>(
        acc, stats, gamma, beta, destH, r.nout, destRows);
  } else {
    long long tot = destRows * COUT;
    bn_apply_f32<<<dim3((unsigned)((tot + 255) / 256)), 256, 0, stream>>>(
        acc, stats, gamma, beta, destF, r.nout, COUT, destRows);
  }
}

extern "C" void kernel_launch(void* const* d_in, const int* in_sizes, int n_in,
                              void* d_out, int out_size, void* d_ws, size_t ws_size,
                              hipStream_t stream) {
  (void)n_in; (void)ws_size;
  const float* vox = (const float*)d_in[0];
  auto Wp = [&](int li) { return (const float*)d_in[1 + li * 3 + 0]; };
  auto Gp = [&](int li) { return (const float*)d_in[1 + li * 3 + 1]; };
  auto Bp = [&](int li) { return (const float*)d_in[1 + li * 3 + 2]; };
  auto mkRule = [&](int ridx, int K) {
    int b = 43 + ridx * 3;
    RuleH r;
    r.rin  = (const int*)d_in[b];
    r.rout = (const int*)d_in[b + 1];
    r.nout = (const int*)d_in[b + 2];
    r.K = K;
    r.L = in_sizes[b] / K;
    return r;
  };
  RuleH subm1  = mkRule(0, 27), spconv2 = mkRule(1, 27), subm2 = mkRule(2, 27);
  RuleH submxy = mkRule(3, 27), spconv3 = mkRule(4, 27), subm3 = mkRule(5, 27);
  RuleH spconv4 = mkRule(6, 27), subm4 = mkRule(7, 27), down2 = mkRule(8, 3);
  const int*   c1 = (const int*)d_in[70];
  const float* vs = (const float*)d_in[71];
  const float* pr = (const float*)d_in[72];

  int n0  = in_sizes[0] / 4;
  int n1  = in_sizes[70] / 4;
  int n4d = (out_size - n1 * 68) / 128;   // out_size = n4d*128 + n1*64 + n1*4

  float* dout = (float*)d_out;

  // ---- 64B-aligned bump allocation from d_ws ----
  char* ws = (char*)d_ws;
  size_t off = 0;
  auto allocB = [&](size_t bytes) {
    void* p = ws + off;
    off += (bytes + 63) & ~(size_t)63;
    return p;
  };
  auto allocH = [&](size_t nh) { return (_Float16*)allocB(nh * 2); };
  auto allocF = [&](size_t nf) { return (float*)allocB(nf * 4); };

  _Float16* F0h  = allocH((size_t)(n0 + 1) * 32);
  _Float16* Ah   = allocH((size_t)(n0 + 1) * 32);
  _Float16* Bh   = allocH((size_t)(n0 + 1) * 32);
  _Float16* X2a  = allocH((size_t)(n1 + 1) * 32);
  _Float16* X2b  = allocH((size_t)(n1 + 1) * 32);
  _Float16* Rh   = allocH((size_t)(n1 + 1) * 64);
  _Float16* Sh   = allocH((size_t)(n1 + 1) * 64);
  _Float16* Ph   = allocH((size_t)(n1 + 1) * 64);
  _Float16* Qh   = allocH((size_t)(n1 + 1) * 64);
  size_t accMax = (size_t)(n0 + 1) * 16;
  if ((size_t)(n1 + 1) * 64 > accMax)  accMax = (size_t)(n1 + 1) * 64;
  if ((size_t)(n4d + 1) * 128 > accMax) accMax = (size_t)(n4d + 1) * 128;
  float*    ACC  = allocF(accMax);
  _Float16* WT   = allocH(131072);     // max 27*64*64 = 110592 halves
  float*    stats = allocF(256);

  // voxel features -> padded half with sentinel row
  {
    long long tot = (long long)(n0 + 1) * 32;
    feat0_prep<<<dim3((unsigned)((tot + 255) / 256)), 256, 0, stream>>>(vox, F0h, n0);
  }

  long long rows0 = (long long)n0 + 1, rows1 = (long long)n1 + 1;

  // ---- backbone ----
  runLayer<4, 16>  (F0h, subm1.nout, subm1,   Wp(0),  Gp(0),  Bp(0),  ACC, stats, WT, n0,  Ah,  nullptr, rows0, stream); // conv_input
  runLayer<16, 16> (Ah,  subm1.nout, subm1,   Wp(1),  Gp(1),  Bp(1),  ACC, stats, WT, n0,  Bh,  nullptr, rows0, stream); // conv1_0
  runLayer<16, 32> (Bh,  subm1.nout, spconv2, Wp(2),  Gp(2),  Bp(2),  ACC, stats, WT, n1,  X2a, nullptr, rows1, stream); // conv2_0
  runLayer<32, 32> (X2a, subm2.nout, subm2,   Wp(3),  Gp(3),  Bp(3),  ACC, stats, WT, n1,  X2b, nullptr, rows1, stream); // conv2_1
  runLayer<32, 32> (X2b, subm2.nout, subm2,   Wp(4),  Gp(4),  Bp(4),  ACC, stats, WT, n1,  X2a, nullptr, rows1, stream); // conv2_2 (x2)
  // points branch
  runLayer<32, 64> (X2a, subm2.nout, submxy,  Wp(11), Gp(11), Bp(11), ACC, stats, WT, n1,  Ph,  nullptr, rows1, stream); // conv_points_0
  runLayer<64, 64> (Ph,  subm2.nout, submxy,  Wp(12), Gp(12), Bp(12), ACC, stats, WT, n1,  nullptr,
                    dout + (size_t)n4d * 128, (long long)n1, stream);                                                     // conv_points_1
  // main branch
  runLayer<32, 64> (X2a, subm2.nout, spconv3, Wp(5),  Gp(5),  Bp(5),  ACC, stats, WT, n1,  Rh,  nullptr, rows1, stream); // conv3_0
  runLayer<64, 64> (Rh,  subm3.nout, subm3,   Wp(6),  Gp(6),  Bp(6),  ACC, stats, WT, n1,  Sh,  nullptr, rows1, stream); // conv3_1
  runLayer<64, 64> (Sh,  subm3.nout, subm3,   Wp(7),  Gp(7),  Bp(7),  ACC, stats, WT, n1,  Rh,  nullptr, rows1, stream); // conv3_2
  runLayer<64, 64> (Rh,  subm3.nout, spconv4, Wp(8),  Gp(8),  Bp(8),  ACC, stats, WT, n1,  Sh,  nullptr, rows1, stream); // conv4_0
  runLayer<64, 64> (Sh,  subm4.nout, subm4,   Wp(9),  Gp(9),  Bp(9),  ACC, stats, WT, n1,  Rh,  nullptr, rows1, stream); // conv4_1
  runLayer<64, 64> (Rh,  subm4.nout, subm4,   Wp(10), Gp(10), Bp(10), ACC, stats, WT, n1,  Sh,  nullptr, rows1, stream); // conv4_2 (x4)
  runLayer<64, 128>(Sh,  subm4.nout, down2,   Wp(13), Gp(13), Bp(13), ACC, stats, WT, n4d, nullptr,
                    dout, (long long)n4d, stream);                                                                        // conv_out

  point_coords_k<<<dim3((unsigned)((n1 + 255) / 256)), 256, 0, stream>>>(
      c1, vs, pr, dout + (size_t)n4d * 128 + (size_t)n1 * 64, n1);
}
